// MIA_SS2D_57878979281532
// MI455X (gfx1250) — compile-verified, matrix operand order checked by reading, not run on hardware
//
#include <hip/hip_runtime.h>
#include <hip/hip_bf16.h>

// ---------------- problem constants ----------------
#define DIMC 96
#define BSZ  2
#define HH   64
#define WW   64
#define LQ   (HH*WW)          // 4096
#define BL   (BSZ*LQ)         // 8192
#define DIN  192              // d_inner
#define NST  16               // d_state
#define RK   6                // dt_rank
#define KD   4                // scan directions
#define QQ   24               // DIM/4
#define XPC  (RK + 2*NST)     // 38 columns of x_proj output
#define KMAX 192              // max GEMM K

// ---------------- vector types ----------------
typedef __attribute__((ext_vector_type(16))) __bf16 bf16x16;
typedef __attribute__((ext_vector_type(8)))  __bf16 bf16x8;
typedef __attribute__((ext_vector_type(8)))  float  f32x8;
typedef unsigned int u32;
typedef u32 __attribute__((ext_vector_type(4))) u32x4;
typedef int __attribute__((ext_vector_type(4))) i32x4;
typedef int __attribute__((ext_vector_type(8))) i32x8;

#if defined(__has_builtin)
#if __has_builtin(__builtin_amdgcn_tensor_load_to_lds) && __has_builtin(__builtin_amdgcn_s_wait_tensorcnt)
#define HAVE_TDM 1
#endif
#endif
#ifndef HAVE_TDM
#define HAVE_TDM 0
#endif

// A/B fragment for v_wmma_f32_16x16x32_bf16, from row-major bf16
// (ISA 7.12.2: lane<16 -> K {kb..kb+7, kb+16..kb+23} with kb=0; lane>=16 kb=8)
__device__ __forceinline__ bf16x16 load_frag(const __bf16* p) {
  bf16x8 lo = *(const bf16x8*)(p);
  bf16x8 hi = *(const bf16x8*)(p + 16);
  bf16x16 r;
#pragma unroll
  for (int i = 0; i < 8; ++i) { r[i] = lo[i]; r[i + 8] = hi[i]; }
  return r;
}

// ---------------- WMMA GEMM: C[M x Nvalid] = A[M x K] * W[Nvalid x K]^T (+bias) ----------------
// grid.x: groups of 8 M-tiles (one per wave); grid.y: 16-column N-strip.
// The block's 16 x K weight strip is staged into LDS once (TDM when available),
// so the K-loop is branch-free: 2x global_load_b128 + 2x ds_load_b128 + v_wmma.
__global__ void k_gemm_bf16(const __hip_bfloat16* __restrict__ A_, int lda,
                            const __hip_bfloat16* __restrict__ W_, int ldw,
                            float* __restrict__ C, int ldc,
                            const float* __restrict__ bias,
                            int M, int Nvalid, int K) {
  __shared__ __align__(16) __bf16 wlds[16 * KMAX];
  const __bf16* A = (const __bf16*)A_;
  const __bf16* W = (const __bf16*)W_;
  int tn = blockIdx.y;
  int rem = Nvalid - tn * 16;   // valid rows in this strip (may be < 16, rest zero)

#if HAVE_TDM
  if (threadIdx.x < 32) {       // one wave issues the TDM descriptor
    unsigned long long ga =
        (unsigned long long)(size_t)(W + (size_t)(tn * 16) * ldw);
    u32 lds_base = (u32)(size_t)&wlds[0];   // flat addr low 32 bits == LDS offset
    int td1 = rem < 16 ? rem : 16;          // OOB rows beyond tensor_dim1 read as 0
    u32x4 g0;
    g0.x = 1u;                               // count=1, user mode, no gather
    g0.y = lds_base;                         // lds_addr [63:32]
    g0.z = (u32)ga;                          // global_addr lo
    g0.w = (u32)(ga >> 32) | (2u << 30);     // global_addr hi | type=2 ("image")
    i32x8 g1;
    g1[0] = 0x00010000;                      // wg_mask=0, data_size=1 (2 bytes)
    g1[1] = (int)((u32)K << 16);             // tensor_dim0 lo16 @ bits[63:48]
    g1[2] = (int)(((u32)td1 & 0xFFFF) << 16);// tensor_dim1 lo16 @ bits[95:80]
    g1[3] = (int)((u32)K << 16);             // tile_dim0 @ bits[127:112]
    g1[4] = 16;                              // tile_dim1 @ bits[143:128]
    g1[5] = K;                               // tensor_dim0_stride lo32
    g1[6] = 0; g1[7] = 0;
    i32x4 gz = {};                           // 2D tensor: groups 2/3 zero
#if __clang_major__ >= 23
    i32x8 gz8 = {};
    __builtin_amdgcn_tensor_load_to_lds(g0, g1, gz, gz, gz8, 0);
#else
    __builtin_amdgcn_tensor_load_to_lds(g0, g1, gz, gz, 0);
#endif
    __builtin_amdgcn_s_wait_tensorcnt(0);
  }
  __syncthreads();
#else
  for (int i = threadIdx.x * 8; i < 16 * K; i += blockDim.x * 8) {
    int row = i / K, col = i - row * K;
    bf16x8 v = {};
    if (row < rem) v = *(const bf16x8*)(W + (size_t)(tn * 16 + row) * ldw + col);
    *(bf16x8*)(&wlds[i]) = v;
  }
  __syncthreads();
#endif

  int tm = blockIdx.x * (blockDim.x >> 5) + (threadIdx.x >> 5);
  if (tm >= (M >> 4)) return;
  int lane = threadIdx.x & 31;
  int half = lane >> 4, l16 = lane & 15;
  int kb = half * 8;
  const __bf16* arow = A + (size_t)(tm * 16 + l16) * lda + kb;
  const __bf16* wrow = &wlds[l16 * K + kb];
  f32x8 acc = {};
  for (int k0 = 0; k0 < K; k0 += 32) {
    bf16x16 af = load_frag(arow + k0);
    bf16x16 bf = load_frag(wrow + k0);     // ds_load_b128 x2 (addrspace inferred)
    acc = __builtin_amdgcn_wmma_f32_16x16x32_bf16(false, af, false, bf,
                                                  (short)0, acc, false, false);
  }
  int col = tn * 16 + l16;
  if (col < Nvalid) {
    float bv = bias ? bias[col] : 0.f;
#pragma unroll
    for (int r = 0; r < 8; ++r)
      C[(size_t)(tm * 16 + r + half * 8) * ldc + col] = acc[r] + bv;
  }
}

// ---------------- fp32 -> bf16 convert ----------------
__global__ void k_tobf16(const float* __restrict__ s, __hip_bfloat16* __restrict__ d, int n) {
  int i = blockIdx.x * blockDim.x + threadIdx.x;
  if (i < n) d[i] = __float2bfloat16(s[i]);
}

// ---------------- ARF: 4 dilated grouped 3x3 convs -> concat, (b,hw,c) bf16 ----------------
__global__ void k_arf(const float* __restrict__ x,
                      const float* __restrict__ w1, const float* __restrict__ b1,
                      const float* __restrict__ w2, const float* __restrict__ b2,
                      const float* __restrict__ w3, const float* __restrict__ b3,
                      const float* __restrict__ b4w, const float* __restrict__ b4b,
                      __hip_bfloat16* __restrict__ cat) {
  int tid = blockIdx.x * blockDim.x + threadIdx.x;
  if (tid >= BSZ * LQ * DIMC) return;
  int c = tid % DIMC, l = (tid / DIMC) % LQ, b = tid / (DIMC * LQ);
  int yy = l / WW, xx = l % WW;
  int branch = c / QQ, oc = c % QQ;
  const float *w, *bb; int dil;
  if      (branch == 0) { w = w1;  bb = b1;  dil = 1; }
  else if (branch == 1) { w = w2;  bb = b2;  dil = 2; }
  else if (branch == 2) { w = w3;  bb = b3;  dil = 4; }
  else                  { w = b4w; bb = b4b; dil = 8; }
  float acc = bb[oc];
  const float* xb = x + (size_t)b * DIMC * LQ;
  for (int ci = 0; ci < 4; ++ci) {
    const float* xc = xb + (size_t)(oc * 4 + ci) * LQ;
    const float* wk = w + (oc * 4 + ci) * 9;
    for (int ky = 0; ky < 3; ++ky) {
      int iy = yy + (ky - 1) * dil;
      if (iy < 0 || iy >= HH) continue;
      for (int kx = 0; kx < 3; ++kx) {
        int ix = xx + (kx - 1) * dil;
        if (ix < 0 || ix >= WW) continue;
        acc += wk[ky * 3 + kx] * xc[iy * WW + ix];
      }
    }
  }
  cat[tid] = __float2bfloat16(acc);
}

// ---------------- depthwise edge conv (given edge_w), h in (b,hw,c) ----------------
__global__ void k_edge(const float* __restrict__ h, const float* __restrict__ ew,
                       float* __restrict__ e) {
  int tid = blockIdx.x * blockDim.x + threadIdx.x;
  if (tid >= BSZ * LQ * DIMC) return;
  int c = tid % DIMC, l = (tid / DIMC) % LQ, b = tid / (DIMC * LQ);
  int yy = l / WW, xx = l % WW;
  const float* hb = h + (size_t)b * LQ * DIMC;
  const float* wk = ew + c * 9;
  float acc = 0.f;
  for (int ky = 0; ky < 3; ++ky) {
    int iy = yy + ky - 1;
    if (iy < 0 || iy >= HH) continue;
    for (int kx = 0; kx < 3; ++kx) {
      int ix = xx + kx - 1;
      if (ix < 0 || ix >= WW) continue;
      acc += wk[ky * 3 + kx] * hb[(size_t)(iy * WW + ix) * DIMC + c];
    }
  }
  e[tid] = acc;
}

// ---------------- enh1: 96 -> 24 (1x1) ----------------
__global__ void k_enh1(const float* __restrict__ edge, const float* __restrict__ w1,
                       const float* __restrict__ b1, float* __restrict__ e) {
  int tid = blockIdx.x * blockDim.x + threadIdx.x;
  if (tid >= BL * QQ) return;
  int j = tid % QQ; int row = tid / QQ;
  const float* r = edge + (size_t)row * DIMC;
  const float* w = w1 + j * DIMC;
  float a = b1[j];
  for (int c = 0; c < DIMC; ++c) a += w[c] * r[c];
  e[tid] = a;
}

// ---------------- BN stats over (B,H,W) per channel (24) ----------------
__global__ void k_bnstats(const float* __restrict__ e, float* __restrict__ mu,
                          float* __restrict__ var) {
  __shared__ float ss[256], sq[256];
  int j = blockIdx.x, t = threadIdx.x;
  float s = 0.f, q = 0.f;
  for (int i = t; i < BL; i += 256) { float v = e[(size_t)i * QQ + j]; s += v; q += v * v; }
  ss[t] = s; sq[t] = q; __syncthreads();
  for (int off = 128; off > 0; off >>= 1) {
    if (t < off) { ss[t] += ss[t + off]; sq[t] += sq[t + off]; }
    __syncthreads();
  }
  if (t == 0) { float m = ss[0] / BL; mu[j] = m; var[j] = sq[0] / BL - m * m; }
}

// ---------------- BN+relu -> enh2 (24->96) -> sigmoid gate on h (in place) ----------------
__global__ void k_enh2_gate(const float* __restrict__ e, const float* __restrict__ mu,
                            const float* __restrict__ var, const float* __restrict__ g,
                            const float* __restrict__ bb, const float* __restrict__ w2,
                            const float* __restrict__ b2, float* __restrict__ h) {
  __shared__ float et[QQ];
  int bl = blockIdx.x, t = threadIdx.x;
  if (t < QQ) {
    float v = (e[(size_t)bl * QQ + t] - mu[t]) * rsqrtf(var[t] + 1e-5f) * g[t] + bb[t];
    et[t] = v > 0.f ? v : 0.f;
  }
  __syncthreads();
  if (t < DIMC) {
    const float* w = w2 + t * QQ;
    float a = b2[t];
    for (int j = 0; j < QQ; ++j) a += w[j] * et[j];
    float sig = 1.f / (1.f + __expf(-a));
    h[(size_t)bl * DIMC + t] *= (1.f + sig);
  }
}

// ---------------- channel recal: s[b,c] = mean_hw + max_hw ----------------
__global__ void k_recal_reduce(const float* __restrict__ h, float* __restrict__ s, int C) {
  __shared__ float su[256], mx[256];
  int b = blockIdx.x / C, c = blockIdx.x % C, t = threadIdx.x;
  float a = 0.f, m = -3.4e38f;
  const float* base = h + (size_t)b * LQ * C + c;
  for (int l = t; l < LQ; l += 256) { float v = base[(size_t)l * C]; a += v; m = v > m ? v : m; }
  su[t] = a; mx[t] = m; __syncthreads();
  for (int off = 128; off > 0; off >>= 1) {
    if (t < off) { su[t] += su[t + off]; mx[t] = mx[t + off] > mx[t] ? mx[t + off] : mx[t]; }
    __syncthreads();
  }
  if (t == 0) s[b * C + c] = su[0] / LQ + mx[0];
}

// ---------------- recal FCs: sigmoid(relu(s@fc1^T)@fc2^T) ----------------
__global__ void k_recal_fc(const float* __restrict__ s, const float* __restrict__ f1,
                           const float* __restrict__ f2, float* __restrict__ wsc) {
  __shared__ float sv[DIMC], tv[QQ];
  int b = blockIdx.x, t = threadIdx.x;
  if (t < DIMC) sv[t] = s[b * DIMC + t];
  __syncthreads();
  if (t < QQ) {
    float a = 0.f;
    for (int c = 0; c < DIMC; ++c) a += f1[t * DIMC + c] * sv[c];
    tv[t] = a > 0.f ? a : 0.f;
  }
  __syncthreads();
  if (t < DIMC) {
    float a = 0.f;
    for (int j = 0; j < QQ; ++j) a += f2[t * QQ + j] * tv[j];
    wsc[b * DIMC + t] = 1.f / (1.f + __expf(-a));
  }
}

// ---------------- scale-by-recal + LayerNorm(96) -> bf16 ----------------
__global__ void k_scale_ln(const float* __restrict__ h, const float* __restrict__ wsc,
                           const float* __restrict__ g, const float* __restrict__ bta,
                           __hip_bfloat16* __restrict__ out) {
  __shared__ float red[128]; __shared__ float stat[2];
  int bl = blockIdx.x, t = threadIdx.x, b = bl / LQ;
  float v = 0.f;
  if (t < DIMC) v = h[(size_t)bl * DIMC + t] * wsc[b * DIMC + t];
  red[t] = v; __syncthreads();
  for (int off = 64; off > 0; off >>= 1) { if (t < off) red[t] += red[t + off]; __syncthreads(); }
  if (t == 0) stat[0] = red[0] / DIMC;
  __syncthreads();
  float mean = stat[0];
  float dv = (t < DIMC) ? (v - mean) : 0.f;
  red[t] = dv * dv; __syncthreads();
  for (int off = 64; off > 0; off >>= 1) { if (t < off) red[t] += red[t + off]; __syncthreads(); }
  if (t == 0) stat[1] = red[0] / DIMC;
  __syncthreads();
  if (t < DIMC)
    out[(size_t)bl * DIMC + t] =
        __float2bfloat16((v - mean) * rsqrtf(stat[1] + 1e-5f) * g[t] + bta[t]);
}

// ---------------- depthwise 3x3 on xm + SiLU -> 4-direction xs (bf16) ----------------
__global__ void k_ssconv_xs(const float* __restrict__ ip, const float* __restrict__ cw,
                            const float* __restrict__ cb, __hip_bfloat16* __restrict__ xs) {
  int tid = blockIdx.x * blockDim.x + threadIdx.x;
  if (tid >= BSZ * LQ * DIN) return;
  int d = tid % DIN, l = (tid / DIN) % LQ, b = tid / (DIN * LQ);
  int yy = l / WW, xx = l % WW;
  const float* base = ip + (size_t)b * LQ * (2 * DIN);
  const float* wk = cw + d * 9;
  float acc = cb[d];
  for (int ky = 0; ky < 3; ++ky) {
    int iy = yy + ky - 1;
    if (iy < 0 || iy >= HH) continue;
    for (int kx = 0; kx < 3; ++kx) {
      int ix = xx + kx - 1;
      if (ix < 0 || ix >= WW) continue;
      acc += wk[ky * 3 + kx] * base[(size_t)(iy * WW + ix) * (2 * DIN) + d];
    }
  }
  float v = acc / (1.f + __expf(-acc));   // SiLU
  __hip_bfloat16 bv = __float2bfloat16(v);
  int lwh = xx * HH + yy;
  size_t xb = (size_t)b * KD * LQ * DIN;
  xs[xb + ((size_t)0 * LQ + l) * DIN + d]              = bv;
  xs[xb + ((size_t)1 * LQ + lwh) * DIN + d]            = bv;
  xs[xb + ((size_t)2 * LQ + (LQ - 1 - l)) * DIN + d]   = bv;
  xs[xb + ((size_t)3 * LQ + (LQ - 1 - lwh)) * DIN + d] = bv;
}

// ---------------- dt projection (rank 6) + softplus ----------------
__global__ void k_dt(const float* __restrict__ xdbl, const float* __restrict__ dtw,
                     const float* __restrict__ dtb, float* __restrict__ dt) {
  int tid = blockIdx.x * blockDim.x + threadIdx.x;
  if (tid >= BSZ * KD * LQ * DIN) return;
  int d = tid % DIN, l = (tid / DIN) % LQ, k = (tid / (DIN * LQ)) % KD;
  int bk = tid / (DIN * LQ);
  const float* xr = xdbl + ((size_t)bk * LQ + l) * XPC;
  const float* w  = dtw + ((size_t)k * DIN + d) * RK;
  float a = dtb[k * DIN + d];
#pragma unroll
  for (int r = 0; r < RK; ++r) a += w[r] * xr[r];
  dt[tid] = (a > 20.f) ? a : log1pf(__expf(a));
}

// ---------------- selective scan: 8 blocks x 192 threads, 16 states/thread ----------------
__global__ void k_scan(const __hip_bfloat16* __restrict__ xs, const float* __restrict__ dt,
                       const float* __restrict__ xdbl, const float* __restrict__ alog,
                       const float* __restrict__ Ds, float* __restrict__ oy) {
  int bk = blockIdx.x;                 // (b*KD + k)
  int k = bk % KD;
  int d = threadIdx.x;                 // 0..191
  float Av[NST], h[NST];
#pragma unroll
  for (int n = 0; n < NST; ++n) {
    Av[n] = -__expf(alog[((size_t)k * DIN + d) * NST + n]);
    h[n] = 0.f;
  }
  float Dv = Ds[k * DIN + d];
  const __hip_bfloat16* xsr = xs + (size_t)bk * LQ * DIN;
  const float* dtr = dt + (size_t)bk * LQ * DIN;
  const float* xdr = xdbl + (size_t)bk * LQ * XPC;
  float* oyr = oy + (size_t)bk * LQ * DIN;
  for (int l = 0; l < LQ; ++l) {
    float dl = dtr[(size_t)l * DIN + d];
    float u  = __bfloat162float(xsr[(size_t)l * DIN + d]);
    float du = dl * u;
    const float* bc = xdr + (size_t)l * XPC;
    if (l + 1 < LQ) __builtin_prefetch(xdr + (size_t)(l + 1) * XPC, 0, 1);
    float y = 0.f;
#pragma unroll
    for (int n = 0; n < NST; ++n) {
      h[n] = __expf(dl * Av[n]) * h[n] + du * bc[RK + n];
      y += h[n] * bc[RK + NST + n];
    }
    oyr[(size_t)l * DIN + d] = y + Dv * u;
  }
}

// ---------------- merge 4 directions + LayerNorm(192) + SiLU(z) gate -> bf16 ----------------
__global__ void k_merge_ln_gate(const float* __restrict__ oy, const float* __restrict__ ip,
                                const float* __restrict__ g, const float* __restrict__ bta,
                                __hip_bfloat16* __restrict__ out) {
  __shared__ float red[DIN]; __shared__ float stat[2];
  int bl = blockIdx.x, b = bl / LQ, l = bl % LQ, t = threadIdx.x;
  int yy = l / WW, xx = l % WW;
  int lwh = xx * HH + yy;
  const float* o = oy + (size_t)b * KD * LQ * DIN;
  float v = o[((size_t)0 * LQ + l) * DIN + t]
          + o[((size_t)2 * LQ + (LQ - 1 - l)) * DIN + t]
          + o[((size_t)1 * LQ + lwh) * DIN + t]
          + o[((size_t)3 * LQ + (LQ - 1 - lwh)) * DIN + t];
  red[t] = v; __syncthreads();
  for (int off = 96; off >= 3; off >>= 1) { if (t < off) red[t] += red[t + off]; __syncthreads(); }
  if (t == 0) stat[0] = (red[0] + red[1] + red[2]) / DIN;
  __syncthreads();
  float mean = stat[0];
  float dv = v - mean;
  red[t] = dv * dv; __syncthreads();
  for (int off = 96; off >= 3; off >>= 1) { if (t < off) red[t] += red[t + off]; __syncthreads(); }
  if (t == 0) stat[1] = (red[0] + red[1] + red[2]) / DIN;
  __syncthreads();
  float ln = dv * rsqrtf(stat[1] + 1e-5f) * g[t] + bta[t];
  float z = ip[((size_t)b * LQ + l) * (2 * DIN) + DIN + t];
  float sz = z / (1.f + __expf(-z));
  out[(size_t)bl * DIN + t] = __float2bfloat16(ln * sz);
}

// ---------------- post-recal scale -> bf16 ----------------
__global__ void k_scale_bf16(const float* __restrict__ hn, const float* __restrict__ wsc,
                             __hip_bfloat16* __restrict__ out) {
  int tid = blockIdx.x * blockDim.x + threadIdx.x;
  if (tid >= BL * DIMC) return;
  int c = tid % DIMC, b = tid / (DIMC * LQ);
  out[tid] = __float2bfloat16(hn[tid] * wsc[b * DIMC + c]);
}

// ---------------- residual + bias + (b,hw,c)->(b,c,h,w) ----------------
__global__ void k_final(const float* __restrict__ x, const float* __restrict__ po,
                        const float* __restrict__ pb, float* __restrict__ out) {
  int tid = blockIdx.x * blockDim.x + threadIdx.x;
  if (tid >= BSZ * DIMC * LQ) return;
  int l = tid % LQ, c = (tid / LQ) % DIMC, b = tid / (LQ * DIMC);
  out[tid] = x[tid] + po[((size_t)b * LQ + l) * DIMC + c] + pb[c];
}

// ================= host launch =================
static inline size_t alup(size_t v) { return (v + 255) & ~size_t(255); }

extern "C" void kernel_launch(void* const* d_in, const int* in_sizes, int n_in,
                              void* d_out, int out_size, void* d_ws, size_t ws_size,
                              hipStream_t stream) {
  (void)in_sizes; (void)n_in; (void)out_size; (void)ws_size;
  const float* x        = (const float*)d_in[0];
  const float* arf_w1   = (const float*)d_in[1];  const float* arf_b1 = (const float*)d_in[2];
  const float* arf_w2   = (const float*)d_in[3];  const float* arf_b2 = (const float*)d_in[4];
  const float* arf_w3   = (const float*)d_in[5];  const float* arf_b3 = (const float*)d_in[6];
  const float* arf_w4   = (const float*)d_in[7];  const float* arf_b4 = (const float*)d_in[8];
  const float* fuse_w   = (const float*)d_in[9];  const float* fuse_b = (const float*)d_in[10];
  const float* edge_w   = (const float*)d_in[11];
  const float* enh_w1   = (const float*)d_in[12]; const float* enh_b1 = (const float*)d_in[13];
  const float* bn_g     = (const float*)d_in[14]; const float* bn_b   = (const float*)d_in[15];
  const float* enh_w2   = (const float*)d_in[16]; const float* enh_b2 = (const float*)d_in[17];
  const float* pre_fc1  = (const float*)d_in[18]; const float* pre_fc2 = (const float*)d_in[19];
  const float* norm_g   = (const float*)d_in[20]; const float* norm_b  = (const float*)d_in[21];
  const float* in_proj_w = (const float*)d_in[22];
  const float* ss_conv_w = (const float*)d_in[23]; const float* ss_conv_b = (const float*)d_in[24];
  const float* x_proj_w  = (const float*)d_in[25];
  const float* dt_w      = (const float*)d_in[26]; const float* dt_b = (const float*)d_in[27];
  const float* A_logs    = (const float*)d_in[28]; const float* Ds   = (const float*)d_in[29];
  const float* out_norm_g = (const float*)d_in[30]; const float* out_norm_b = (const float*)d_in[31];
  const float* out_proj_w = (const float*)d_in[32];
  const float* post_fc1  = (const float*)d_in[33]; const float* post_fc2 = (const float*)d_in[34];
  const float* proj_w    = (const float*)d_in[35]; const float* proj_b   = (const float*)d_in[36];

  char* ws = (char*)d_ws;
  size_t off = 0;
  auto alloc = [&](size_t bytes) { void* p = ws + off; off = alup(off + bytes); return p; };

  __hip_bfloat16* catb = (__hip_bfloat16*)alloc((size_t)BL * DIMC * 2);
  float*          hbuf = (float*)alloc((size_t)BL * DIMC * 4);
  float*          edg  = (float*)alloc((size_t)BL * DIMC * 4);
  float*          ebuf = (float*)alloc((size_t)BL * QQ * 4);
  float*          mu   = (float*)alloc(QQ * 4);
  float*          var  = (float*)alloc(QQ * 4);
  float*          sred = (float*)alloc(BSZ * DIMC * 4);
  float*          wsc  = (float*)alloc(BSZ * DIMC * 4);
  __hip_bfloat16* lnb  = (__hip_bfloat16*)alloc((size_t)BL * DIMC * 2);
  float*          ip   = (float*)alloc((size_t)BL * 2 * DIN * 4);
  __hip_bfloat16* xsb  = (__hip_bfloat16*)alloc((size_t)BSZ * KD * LQ * DIN * 2);
  float*          xdbl = (float*)alloc((size_t)BSZ * KD * LQ * XPC * 4);
  float*          dtv  = (float*)alloc((size_t)BSZ * KD * LQ * DIN * 4);
  float*          oy   = (float*)alloc((size_t)BSZ * KD * LQ * DIN * 4);
  __hip_bfloat16* ymg  = (__hip_bfloat16*)alloc((size_t)BL * DIN * 2);
  float*          hn   = (float*)alloc((size_t)BL * DIMC * 4);
  __hip_bfloat16* prin = (__hip_bfloat16*)alloc((size_t)BL * DIMC * 2);
  float*          po   = (float*)alloc((size_t)BL * DIMC * 4);
  __hip_bfloat16* fuse_wb = (__hip_bfloat16*)alloc(DIMC * DIMC * 2);
  __hip_bfloat16* inpj_wb = (__hip_bfloat16*)alloc((size_t)2 * DIN * DIMC * 2);
  __hip_bfloat16* xprj_wb = (__hip_bfloat16*)alloc((size_t)KD * XPC * DIN * 2);
  __hip_bfloat16* oprj_wb = (__hip_bfloat16*)alloc((size_t)DIMC * DIN * 2);
  __hip_bfloat16* proj_wb = (__hip_bfloat16*)alloc(DIMC * DIMC * 2);

  auto cvt = [&](const float* s, __hip_bfloat16* d, int n) {
    k_tobf16<<<(n + 255) / 256, 256, 0, stream>>>(s, d, n);
  };
  cvt(fuse_w,     fuse_wb, DIMC * DIMC);
  cvt(in_proj_w,  inpj_wb, 2 * DIN * DIMC);
  cvt(x_proj_w,   xprj_wb, KD * XPC * DIN);
  cvt(out_proj_w, oprj_wb, DIMC * DIN);
  cvt(proj_w,     proj_wb, DIMC * DIMC);

  auto gemm = [&](const __hip_bfloat16* A, int lda, const __hip_bfloat16* W, int ldw,
                  float* C, int ldc, const float* bias, int M, int Nt, int Nv, int K) {
    dim3 grid(((M >> 4) + 7) / 8, Nt);
    k_gemm_bf16<<<grid, 256, 0, stream>>>(A, lda, W, ldw, C, ldc, bias, M, Nv, K);
  };

  // front: ARF -> fuse (WMMA) -> edge -> enh1 -> BN -> gate
  k_arf<<<(BL * DIMC + 255) / 256, 256, 0, stream>>>(x, arf_w1, arf_b1, arf_w2, arf_b2,
                                                     arf_w3, arf_b3, arf_w4, arf_b4, catb);
  gemm(catb, DIMC, fuse_wb, DIMC, hbuf, DIMC, fuse_b, BL, DIMC / 16, DIMC, DIMC);
  k_edge<<<(BL * DIMC + 255) / 256, 256, 0, stream>>>(hbuf, edge_w, edg);
  k_enh1<<<(BL * QQ + 255) / 256, 256, 0, stream>>>(edg, enh_w1, enh_b1, ebuf);
  k_bnstats<<<QQ, 256, 0, stream>>>(ebuf, mu, var);
  k_enh2_gate<<<BL, 128, 0, stream>>>(ebuf, mu, var, bn_g, bn_b, enh_w2, enh_b2, hbuf);

  // pre channel recal + LN -> bf16
  k_recal_reduce<<<BSZ * DIMC, 256, 0, stream>>>(hbuf, sred, DIMC);
  k_recal_fc<<<BSZ, 128, 0, stream>>>(sred, pre_fc1, pre_fc2, wsc);
  k_scale_ln<<<BL, 128, 0, stream>>>(hbuf, wsc, norm_g, norm_b, lnb);

  // SS2D: in_proj (WMMA) -> dwconv+SiLU -> xs -> x_proj (WMMA x8) -> dt -> scan
  gemm(lnb, DIMC, inpj_wb, DIMC, ip, 2 * DIN, nullptr, BL, (2 * DIN) / 16, 2 * DIN, DIMC);
  k_ssconv_xs<<<(BL * DIN + 255) / 256, 256, 0, stream>>>(ip, ss_conv_w, ss_conv_b, xsb);
  for (int bk = 0; bk < BSZ * KD; ++bk) {
    int k = bk % KD;
    gemm(xsb + (size_t)bk * LQ * DIN, DIN, xprj_wb + (size_t)k * XPC * DIN, DIN,
         xdbl + (size_t)bk * LQ * XPC, XPC, nullptr, LQ, 3 /* 48/16 */, XPC, DIN);
  }
  k_dt<<<(BSZ * KD * LQ * DIN + 255) / 256, 256, 0, stream>>>(xdbl, dt_w, dt_b, dtv);
  k_scan<<<BSZ * KD, DIN, 0, stream>>>(xsb, dtv, xdbl, A_logs, Ds, oy);

  // merge + out-LN + SiLU(z) gate -> out_proj (WMMA)
  k_merge_ln_gate<<<BL, DIN, 0, stream>>>(oy, ip, out_norm_g, out_norm_b, ymg);
  gemm(ymg, DIN, oprj_wb, DIN, hn, DIMC, nullptr, BL, DIMC / 16, DIMC, DIN);

  // post channel recal -> proj (WMMA) -> residual
  k_recal_reduce<<<BSZ * DIMC, 256, 0, stream>>>(hn, sred, DIMC);
  k_recal_fc<<<BSZ, 128, 0, stream>>>(sred, post_fc1, post_fc2, wsc);
  k_scale_bf16<<<(BL * DIMC + 255) / 256, 256, 0, stream>>>(hn, wsc, prin);
  gemm(prin, DIMC, proj_wb, DIMC, po, DIMC, nullptr, BL, DIMC / 16, DIMC, DIMC);
  k_final<<<(BSZ * DIMC * LQ + 255) / 256, 256, 0, stream>>>(x, po, proj_b, (float*)d_out);
}
